// MultiGATLayer_52424370815427
// MI455X (gfx1250) — matrix operand order repaired and checked by reading, not compiled, not run on hardware
//
#include <hip/hip_runtime.h>

#define NN   8192
#define DEG  16
#define FIN  1024
#define FOUT 128
#define NH   8

typedef _Float16 v16h __attribute__((ext_vector_type(16)));
typedef _Float16 v4h  __attribute__((ext_vector_type(4)));
typedef float    v8f  __attribute__((ext_vector_type(8)));

// ---------------------------------------------------------------------------
// Kernel 1: hbuf[h][n][o] = features[n] @ W[h] + bW[h]   (f16 WMMA, f32 accum)
// Block = 256 threads (8 wave32), tile = 128(M) x 128(N=FOUT), K-step = 32.
// LDS holds A/B in WMMA-fragment-major layout so each lane loads its v16h
// fragment contiguously (2 x ds_load_b128). Double-buffered.
// ---------------------------------------------------------------------------
__global__ __launch_bounds__(256) void gat_gemm(
    const float* __restrict__ feat, const float* __restrict__ W,
    const float* __restrict__ bW, float* __restrict__ hbuf)
{
    const int h    = blockIdx.y;
    const int n0   = blockIdx.x * 128;
    const int tid  = threadIdx.x;
    const int lane = tid & 31;
    const int wv   = tid >> 5;
    const int wr   = wv >> 1;     // 0..3 : 32-row strip
    const int wc   = wv & 1;      // 0..1 : 64-col strip

    __shared__ __align__(32) _Float16 lA[2][128 * 32];   // 8 KB each buf
    __shared__ __align__(32) _Float16 lB[2][32 * 128];   // 8 KB each buf

    const float* Wh = W + (size_t)h * FIN * FOUT;

    float4 stA[4];       // staged global A (16 f32 / thread)
    float  stB[4][4];    // staged global B (16 f32 / thread)

    // ---- global -> register staging ----
    auto loadA = [&](int ks) {
        const int k0 = ks * 32;
        const int r0 = tid >> 3;          // 0..31
        const int c0 = (tid & 7) * 4;     // 0,4,...,28
        #pragma unroll
        for (int i = 0; i < 4; i++) {
            const int m = r0 + i * 32;
            stA[i] = *(const float4*)&feat[(size_t)(n0 + m) * FIN + k0 + c0];
        }
    };
    auto loadB = [&](int ks) {
        const int k0 = ks * 32;
        const int c  = tid & 127;
        const int g0 = tid >> 7;          // 0..1
        #pragma unroll
        for (int i = 0; i < 4; i++) {
            const int g = g0 + i * 2;     // 0..7, 4 K-rows each
            #pragma unroll
            for (int j = 0; j < 4; j++)
                stB[i][j] = Wh[(size_t)(k0 + g * 4 + j) * FOUT + c];
        }
    };

    // ---- register -> LDS (cvt f32->f16, fragment-major placement) ----
    // A 16x32 f16 frag, lane L: M=L&15, K = 8*(L>=16) + 16*(e>>3) + (e&7)
    auto storeA = [&](int buf) {
        const int r0     = tid >> 3;
        const int c0     = (tid & 7) * 4;
        const int laneHi = (c0 >> 3) & 1;
        const int e0     = ((c0 >> 4) << 3) | (c0 & 7);
        #pragma unroll
        for (int i = 0; i < 4; i++) {
            const int m  = r0 + i * 32;
            const int mi = m >> 4;
            const int ln = (m & 15) | (laneHi << 4);
            v4h v;
            v.x = (_Float16)stA[i].x; v.y = (_Float16)stA[i].y;
            v.z = (_Float16)stA[i].z; v.w = (_Float16)stA[i].w;
            *(v4h*)&lA[buf][((mi * 32 + ln) << 4) + e0] = v;
        }
    };
    // B 32x16 f16 frag, lane L: N=L&15, K = 8*(L>=16) + 16*(e>>3) + (e&7)
    auto storeB = [&](int buf) {
        const int c  = tid & 127;
        const int g0 = tid >> 7;
        const int ni = c >> 4;
        #pragma unroll
        for (int i = 0; i < 4; i++) {
            const int g  = g0 + i * 2;
            const int ln = (c & 15) | (((g >> 1) & 1) << 4);
            const int e0 = ((g >> 2) << 3) | ((g & 1) << 2);
            v4h v;
            v.x = (_Float16)stB[i][0]; v.y = (_Float16)stB[i][1];
            v.z = (_Float16)stB[i][2]; v.w = (_Float16)stB[i][3];
            *(v4h*)&lB[buf][((ni * 32 + ln) << 4) + e0] = v;
        }
    };

    v8f acc[2][4];
    {
        v8f z = {};
        #pragma unroll
        for (int i = 0; i < 2; i++)
            #pragma unroll
            for (int j = 0; j < 4; j++) acc[i][j] = z;
    }

    loadA(0); loadB(0);
    storeA(0); storeB(0);

    const int NK = FIN / 32;      // 32 K-steps
    int cur = 0;
    for (int ks = 0; ks < NK; ks++) {
        __syncthreads();
        if (ks + 1 < NK) { loadA(ks + 1); loadB(ks + 1); }

        v16h af[2], bf[4];
        #pragma unroll
        for (int i = 0; i < 2; i++) {
            const int mi = wr * 2 + i;
            af[i] = *(const v16h*)&lA[cur][(mi * 32 + lane) << 4];
        }
        #pragma unroll
        for (int j = 0; j < 4; j++) {
            const int ni = wc * 4 + j;
            bf[j] = *(const v16h*)&lB[cur][(ni * 32 + lane) << 4];
        }
        #pragma unroll
        for (int i = 0; i < 2; i++)
            #pragma unroll
            for (int j = 0; j < 4; j++)
                acc[i][j] = __builtin_amdgcn_wmma_f32_16x16x32_f16(
                    false, af[i], false, bf[j], (short)0, acc[i][j],
                    false, false);

        if (ks + 1 < NK) { storeA(cur ^ 1); storeB(cur ^ 1); cur ^= 1; }
    }

    // ---- epilogue: +bW, store f32. C/D layout: VGPR r -> M=r (+8 if lane>=16)
    #pragma unroll
    for (int i = 0; i < 2; i++) {
        const int mi = wr * 2 + i;
        const int rowBase = mi * 16 + ((lane >> 4) << 3);
        #pragma unroll
        for (int j = 0; j < 4; j++) {
            const int ni  = wc * 4 + j;
            const int col = ni * 16 + (lane & 15);
            const float b = bW[h * FOUT + col];
            #pragma unroll
            for (int r = 0; r < 8; r++) {
                const int node = n0 + rowBase + r;
                hbuf[((size_t)h * NN + node) * FOUT + col] = acc[i][j][r] + b;
            }
        }
    }
}

// ---------------------------------------------------------------------------
// Kernel 2: s_dst[h,n] = h[h,n,:]·a_dst[h], s_src[h,n] = h[h,n,:]·a_src[h]
// One wave32 per (h,n); lane handles 4 features; shfl_xor tree reduction.
// ---------------------------------------------------------------------------
__global__ __launch_bounds__(256) void gat_scores(
    const float* __restrict__ hbuf, const float* __restrict__ a,
    float* __restrict__ sdst, float* __restrict__ ssrc)
{
    const int lane = threadIdx.x & 31;
    const int wv   = threadIdx.x >> 5;
    const int idx  = blockIdx.x * 8 + wv;       // = h*NN + n  (NN = 2^13)
    const int h    = idx >> 13;

    const float4 hv = *(const float4*)&hbuf[(size_t)idx * FOUT + lane * 4];
    const float4 ad = *(const float4*)&a[h * 2 * FOUT + lane * 4];
    const float4 as = *(const float4*)&a[h * 2 * FOUT + FOUT + lane * 4];

    float d = hv.x * ad.x + hv.y * ad.y + hv.z * ad.z + hv.w * ad.w;
    float s = hv.x * as.x + hv.y * as.y + hv.z * as.z + hv.w * as.w;
    #pragma unroll
    for (int off = 16; off >= 1; off >>= 1) {
        d += __shfl_xor(d, off, 32);
        s += __shfl_xor(s, off, 32);
    }
    if (lane == 0) { sdst[idx] = d; ssrc[idx] = s; }
}

// ---------------------------------------------------------------------------
// Kernel 3: leaky-relu + softmax over 17 neighbors, then weighted gather-sum.
// One 128-thread block per (n, h); thread t owns output feature t.
// ---------------------------------------------------------------------------
__global__ __launch_bounds__(128) void gat_attend(
    const float* __restrict__ hbuf, const int* __restrict__ adj,
    const float* __restrict__ sdst, const float* __restrict__ ssrc,
    const float* __restrict__ ba, float* __restrict__ out)
{
    const int n = blockIdx.x;
    const int h = blockIdx.y;
    const int t = threadIdx.x;

    __shared__ float sc[17];
    __shared__ int   nb[17];

    if (t < 17) {
        const int nv = (t < DEG) ? adj[n * DEG + t] : n;
        nb[t] = nv;
        float x = sdst[h * NN + n] + ssrc[h * NN + nv] + ba[h];
        sc[t] = (x >= 0.f) ? x : 0.2f * x;       // leaky_relu(0.2)
    }
    __syncthreads();

    float mx = -3.402823466e38f;
    #pragma unroll
    for (int d = 0; d < 17; d++) mx = fmaxf(mx, sc[d]);
    float e[17];
    float sum = 0.f;
    #pragma unroll
    for (int d = 0; d < 17; d++) { e[d] = __expf(sc[d] - mx); sum += e[d]; }
    const float inv = 1.f / sum;

    float acc = 0.f;
    #pragma unroll
    for (int d = 0; d < 17; d++)
        acc += e[d] * hbuf[((size_t)h * NN + nb[d]) * FOUT + t];

    out[(size_t)n * (NH * FOUT) + h * FOUT + t] = acc * inv;
}

// ---------------------------------------------------------------------------
extern "C" void kernel_launch(void* const* d_in, const int* in_sizes, int n_in,
                              void* d_out, int out_size, void* d_ws, size_t ws_size,
                              hipStream_t stream)
{
    const float* feat = (const float*)d_in[0];   // (N, F_IN)
    const int*   adj  = (const int*)  d_in[1];   // (N, DEG)
    const float* W    = (const float*)d_in[2];   // (H, F_IN, F_OUT)
    const float* bW   = (const float*)d_in[3];   // (H, F_OUT)
    const float* a    = (const float*)d_in[4];   // (H, 2*F_OUT)
    const float* ba   = (const float*)d_in[5];   // (H,)
    float* out = (float*)d_out;

    float* hbuf = (float*)d_ws;                                       // 32 MB
    float* sdst = (float*)((char*)d_ws + (size_t)NH * NN * FOUT * 4); // 256 KB
    float* ssrc = sdst + NH * NN;                                     // 256 KB

    gat_gemm  <<<dim3(NN / 128, NH), 256, 0, stream>>>(feat, W, bW, hbuf);
    gat_scores<<<NH * NN / 8,        256, 0, stream>>>(hbuf, a, sdst, ssrc);
    gat_attend<<<dim3(NN, NH),       128, 0, stream>>>(hbuf, adj, sdst, ssrc, ba, out);
}
